// LayerBlock_32212254720480
// MI455X (gfx1250) — matrix operand order repaired and checked by reading, not compiled, not run on hardware
//
#include <hip/hip_runtime.h>
#include <hip/hip_bf16.h>
#include <cstdint>

#define DEV static __device__ __forceinline__

typedef __attribute__((ext_vector_type(16))) __bf16 v16bf;
typedef __attribute__((ext_vector_type(8)))  float  v8f;
typedef long long LL;

namespace {
constexpr int   Bb = 2, Cc = 128, Nn = 2000, HEADS = 4, HDim = 32;
constexpr int   MC = 128;       // cluster rows (M in reference)
constexpr int   TOPM = 128, KNNk = 8;
constexpr float BN_EPS = 1e-5f, IN_EPS = 1e-3f;
constexpr float SM_SCALE = 0.17677669529663687f; // 1/sqrt(HD)
}

// ---------------- bf16 conversion (RNE) ----------------
DEV __bf16 f2bf(float x) {
  union { float f; uint32_t u; } a; a.f = x;
  uint32_t r = a.u + 0x7FFFu + ((a.u >> 16) & 1u);
  union { uint16_t s; __bf16 b; } o; o.s = (uint16_t)(r >> 16);
  return o.b;
}

// ---------------- async global->LDS (CDNA5) ----------------
#if __has_builtin(__builtin_amdgcn_global_load_async_to_lds_b32) && \
    __has_builtin(__builtin_amdgcn_s_wait_asynccnt)
#define ATHENA_ASYNC_LDS 1
typedef __attribute__((address_space(1))) int glob_i;
typedef __attribute__((address_space(3))) int lds_i;
DEV void async_g2l(const float* gp, float* lp) {
  __builtin_amdgcn_global_load_async_to_lds_b32(
      (glob_i*)(unsigned long long)(uintptr_t)gp,
      (lds_i*)(unsigned)(uintptr_t)lp, 0, 0);
}
DEV void async_wait0() { __builtin_amdgcn_s_wait_asynccnt(0); }
#endif

// ---------------- Tensor Data Mover (CDNA5 TDM) ----------------
// D# descriptor per ISA ch.8: group0 = {flags, lds_addr, global_addr, type},
// group1 = {mask/data_size/pad, tensor dims, tile dims, dim0 stride}.
// 2D tile load with LDS row padding (pad 1 DWORD every 2^(code+1) DWORDs).
#if __has_builtin(__builtin_amdgcn_tensor_load_to_lds) && \
    __has_builtin(__builtin_amdgcn_s_wait_tensorcnt)
#define ATHENA_TDM 1
typedef __attribute__((ext_vector_type(4))) unsigned int u32x4;
typedef __attribute__((ext_vector_type(8))) int i32x8;
typedef __attribute__((ext_vector_type(4))) int i32x4;
DEV void tdm_load_2d(const float* gbase, void* ldsdst,
                     unsigned tensor_d0, unsigned tensor_d1,
                     unsigned tile_d0, unsigned tile_d1,
                     unsigned long long stride0_elems,
                     unsigned pad_code) {
  unsigned long long ga = (unsigned long long)(uintptr_t)gbase;
  unsigned lds = (unsigned)(uintptr_t)ldsdst;
  u32x4 g0;
  g0[0] = 1u;                                              // count=1 (valid user D#)
  g0[1] = lds;                                             // lds_addr
  g0[2] = (unsigned)(ga & 0xFFFFFFFFu);                    // global_addr[31:0]
  g0[3] = (unsigned)((ga >> 32) & 0x01FFFFFFu) | 0x80000000u; // addr[56:32] | type=2
  i32x8 g1;
  g1[0] = (int)((2u << 16) | (1u << 20) | (pad_code << 22)); // data_size=4B, pad_en, pad=1 DWORD
  g1[1] = (int)((tensor_d0 & 0xFFFFu) << 16);              // tensor_dim0[15:0] @ bit48
  g1[2] = (int)((tensor_d0 >> 16) | ((tensor_d1 & 0xFFFFu) << 16));
  g1[3] = (int)((tensor_d1 >> 16) | (tile_d0 << 16));      // tile_dim0 @ bit112
  g1[4] = (int)(tile_d1 & 0xFFFFu);                        // tile_dim1 @ bit128, tile_dim2=0
  g1[5] = (int)(unsigned)(stride0_elems & 0xFFFFFFFFu);    // tensor_dim0_stride @ bit160
  g1[6] = (int)(unsigned)((stride0_elems >> 32) & 0xFFFFu);
  g1[7] = 0;
  i32x4 z4 = {0, 0, 0, 0};
  i32x8 z8 = {0, 0, 0, 0, 0, 0, 0, 0};
  __builtin_amdgcn_tensor_load_to_lds(g0, g1, z4, z4, z8, 0);
}
DEV void tdm_wait0() { __builtin_amdgcn_s_wait_tensorcnt(0); }
#endif

DEV void atomicMaxF(float* a, float v) {
  unsigned* ai = (unsigned*)a;
  unsigned old = *ai;
  for (;;) {
    float of = __uint_as_float(old);
    if (of >= v) break;
    unsigned assumed = old;
    old = atomicCAS(ai, assumed, __float_as_uint(v));
    if (old == assumed) break;
  }
}

// =====================================================================
// Generic strided batched GEMM: D[g](m,n) = sum_k A[g](m,k)*B[g](k,n)
// 4 waves/block, each wave one 16x16 WMMA tile (v_wmma_f32_16x16x32_bf16)
// Tiles staged via TDM (contiguous operands) or async global->LDS.
// =====================================================================
struct GemmP {
  const float* A; const float* B; float* D;
  const float* bias; const float* res;
  LL sAg, sAm, sAk;
  LL sBg, sBk, sBn;
  LL sDg, sDm, sDn;
  LL sRg, sRm, sRn;
  int M, N, K, relu;
};

__global__ __launch_bounds__(128) void k_gemm(GemmP p) {
  __shared__ float As[64][33];   // 32 DWORDs + 1 pad per row (TDM pad_code=4)
  __shared__ float Bs[32][17];   // 16 DWORDs + 1 pad per row (TDM pad_code=3)
  const int g = blockIdx.z;
  const int n0 = blockIdx.x * 16;
  const int mBlk = blockIdx.y * 64;
  const int t = threadIdx.x;
  const int lane = t & 31, wave = t >> 5;
  const int mW = mBlk + wave * 16;
  const float* Ag = p.A + (LL)g * p.sAg;
  const float* Bg = p.B + (LL)g * p.sBg;
  v8f acc = {0.f, 0.f, 0.f, 0.f, 0.f, 0.f, 0.f, 0.f};
  const int ksteps = (p.K + 31) >> 5;
#ifdef ATHENA_TDM
  const bool tdmA = (p.sAk == 1);
  const bool tdmB = (p.sBn == 1);
#else
  const bool tdmA = false, tdmB = false;
#endif
  for (int ks = 0; ks < ksteps; ++ks) {
    const int kk = ks << 5;
    // ---- stage A tile (64x32 f32) ----
    if (tdmA) {
#ifdef ATHENA_TDM
      if (wave == 0) {
        int remK = p.K - kk; if (remK < 0) remK = 0;
        int remM = p.M - mBlk; if (remM < 0) remM = 0;
        tdm_load_2d(Ag + (LL)mBlk * p.sAm + kk, &As[0][0],
                    (unsigned)remK, (unsigned)remM, 32u, 64u,
                    (unsigned long long)p.sAm, 4u);
      }
#endif
    } else {
      for (int i = t; i < 64 * 32; i += 128) {
        int m = i >> 5, k2 = i & 31;
        int mg = mBlk + m, kg = kk + k2;
        bool inb = (mg < p.M) && (kg < p.K);
#ifdef ATHENA_ASYNC_LDS
        if (inb) async_g2l(&Ag[(LL)mg * p.sAm + (LL)kg * p.sAk], &As[m][k2]);
        else     As[m][k2] = 0.f;
#else
        As[m][k2] = inb ? Ag[(LL)mg * p.sAm + (LL)kg * p.sAk] : 0.f;
#endif
      }
    }
    // ---- stage B tile (32x16 f32) ----
    if (tdmB) {
#ifdef ATHENA_TDM
      if (wave == 1) {
        int remK = p.K - kk; if (remK < 0) remK = 0;
        int remN = p.N - n0; if (remN < 0) remN = 0;
        tdm_load_2d(Bg + (LL)kk * p.sBk + n0, &Bs[0][0],
                    (unsigned)remN, (unsigned)remK, 16u, 32u,
                    (unsigned long long)p.sBk, 3u);
      }
#endif
    } else {
      for (int i = t; i < 32 * 16; i += 128) {
        int k2 = i >> 4, n = i & 15;
        int kg = kk + k2, ng = n0 + n;
        bool inb = (kg < p.K) && (ng < p.N);
#ifdef ATHENA_ASYNC_LDS
        if (inb) async_g2l(&Bg[(LL)kg * p.sBk + (LL)ng * p.sBn], &Bs[k2][n]);
        else     Bs[k2][n] = 0.f;
#else
        Bs[k2][n] = inb ? Bg[(LL)kg * p.sBk + (LL)ng * p.sBn] : 0.f;
#endif
      }
    }
    // prefetch next A tile into cache hierarchy
    if (ks + 1 < ksteps) {
      int mg = mBlk + (t & 63);
      int kg2 = kk + 32 + ((t >> 6) ? 16 : 0);
      if (mg < p.M && kg2 < p.K)
        __builtin_prefetch(&Ag[(LL)mg * p.sAm + (LL)kg2 * p.sAk], 0, 1);
    }
#ifdef ATHENA_TDM
    if ((tdmA && wave == 0) || (tdmB && wave == 1)) tdm_wait0();
#endif
#ifdef ATHENA_ASYNC_LDS
    async_wait0();
#endif
    __syncthreads();
    if (mW < p.M) {
      v16bf a, b;
      const int mr = lane & 15, half = lane >> 4;
#pragma unroll
      for (int j = 0; j < 4; ++j) {
        int k2 = half * 8 + 2 * j;
        a[2 * j]     = f2bf(As[wave * 16 + mr][k2]);
        a[2 * j + 1] = f2bf(As[wave * 16 + mr][k2 + 1]);
        a[2 * j + 8] = f2bf(As[wave * 16 + mr][16 + k2]);
        a[2 * j + 9] = f2bf(As[wave * 16 + mr][16 + k2 + 1]);
      }
#pragma unroll
      for (int j = 0; j < 8; ++j) {
        int k2 = half * 8 + 2 * j + (j >= 4 ? 8 : 0);
        b[2 * j]     = f2bf(Bs[k2][mr]);
        b[2 * j + 1] = f2bf(Bs[k2 + 1][mr]);
      }
      acc = __builtin_amdgcn_wmma_f32_16x16x32_bf16(false, a, false, b,
                                                    (short)0, acc, false, false);
    }
    __syncthreads();
  }
  if (mW < p.M) {
    const int nIdx = lane & 15, half = lane >> 4;
    float* Dg = p.D + (LL)g * p.sDg;
#pragma unroll
    for (int r = 0; r < 8; ++r) {
      int m = mW + r + half * 8;
      int n = n0 + nIdx;
      if (m < p.M && n < p.N) {
        float v = acc[r];
        if (p.bias) v += p.bias[m];
        if (p.res)  v += p.res[(LL)g * p.sRg + (LL)m * p.sRm + (LL)n * p.sRn];
        if (p.relu) v = fmaxf(v, 0.f);
        Dg[(LL)m * p.sDm + (LL)n * p.sDn] = v;
      }
    }
  }
}

// =====================================================================
// Fused local-context conv: WMMA over 2C with on-the-fly KNN gather,
// folded IN2d+BN2d scale/shift + ReLU on B operand, max over k epilogue.
// grid = (TOPM*KNN/16, Cout/16, g) ; one wave per block.
// =====================================================================
__global__ __launch_bounds__(32) void k_localconv(
    const float* __restrict__ W, const float* __restrict__ bw,
    const float* __restrict__ xt, const int* __restrict__ nn,
    const float* __restrict__ scl, const float* __restrict__ sh,
    float* __restrict__ dnew) {
  __shared__ float As[16][33];
  __shared__ float Bs[32][17];
  __shared__ float sacc[8][32];
  const int g = blockIdx.z, o0 = blockIdx.y * 16, nt = blockIdx.x;
  const int lane = threadIdx.x;
  v8f acc = {0.f, 0.f, 0.f, 0.f, 0.f, 0.f, 0.f, 0.f};
  const float* xg = xt + (LL)g * TOPM * Cc;
  const int* ng = nn + (LL)g * TOPM * KNNk;
  const float* sc = scl + (LL)g * 2 * Cc;
  const float* shf = sh + (LL)g * 2 * Cc;
  for (int ch0 = 0; ch0 < 2 * Cc; ch0 += 32) {
    // weight tile via TDM (rows contiguous), gathered B tile via VALU
#ifdef ATHENA_TDM
    tdm_load_2d(W + (LL)(o0) * 2 * Cc + ch0, &As[0][0],
                (unsigned)(2 * Cc - ch0), (unsigned)(Cc - o0), 32u, 16u,
                (unsigned long long)(2 * Cc), 4u);
#else
    for (int i = lane; i < 16 * 32; i += 32) {
      int m = i >> 5, k2 = i & 31;
      As[m][k2] = W[(o0 + m) * 2 * Cc + ch0 + k2];
    }
#endif
    for (int i = lane; i < 32 * 16; i += 32) {
      int k2 = i >> 4, n = i & 15;
      int ch = ch0 + k2;
      int mi = nt * 2 + (n >> 3);
      int kq = n & 7;
      int chl = ch & (Cc - 1);
      float cen = xg[mi * Cc + chl];
      float v = (ch < Cc) ? cen : (cen - xg[ng[mi * KNNk + kq] * Cc + chl]);
      v = v * sc[ch] + shf[ch];
      Bs[k2][n] = fmaxf(v, 0.f);
    }
#ifdef ATHENA_TDM
    tdm_wait0();
#endif
    __syncthreads();
    v16bf a, b;
    const int mr = lane & 15, half = lane >> 4;
#pragma unroll
    for (int j = 0; j < 4; ++j) {
      int k2 = half * 8 + 2 * j;
      a[2 * j]     = f2bf(As[mr][k2]);
      a[2 * j + 1] = f2bf(As[mr][k2 + 1]);
      a[2 * j + 8] = f2bf(As[mr][16 + k2]);
      a[2 * j + 9] = f2bf(As[mr][16 + k2 + 1]);
    }
#pragma unroll
    for (int j = 0; j < 8; ++j) {
      int k2 = half * 8 + 2 * j + (j >= 4 ? 8 : 0);
      b[2 * j]     = f2bf(Bs[k2][mr]);
      b[2 * j + 1] = f2bf(Bs[k2 + 1][mr]);
    }
    acc = __builtin_amdgcn_wmma_f32_16x16x32_bf16(false, a, false, b,
                                                  (short)0, acc, false, false);
    __syncthreads();
  }
  const int half = lane >> 4;
#pragma unroll
  for (int r = 0; r < 8; ++r) sacc[r][lane] = acc[r] + bw[o0 + r + half * 8];
  __syncthreads();
  int rr = lane & 7, grp = lane >> 3;
  int h2 = grp >> 1, ml = grp & 1;
  float mx = -3.4e38f;
#pragma unroll
  for (int tq = 0; tq < 8; ++tq) mx = fmaxf(mx, sacc[rr][h2 * 16 + ml * 8 + tq]);
  int o = o0 + rr + h2 * 8;
  int mcol = nt * 2 + ml;
  dnew[((LL)g * Cc + o) * TOPM + mcol] = mx;
}

// ---------------- reductions / normalization glue ----------------
__global__ void k_stats(const float* __restrict__ X, float* mu, float* rstd,
                        LL sg, int L0, LL s0, int L1, LL s1, float eps) {
  __shared__ float ss[256], sq[256];
  const int g = blockIdx.x, t = threadIdx.x;
  const float* Xg = X + (LL)g * sg;
  const LL total = (LL)L0 * L1;
  float s = 0.f, q = 0.f;
  for (LL i = t; i < total; i += 256) {
    int i0 = (int)(i / L1), i1 = (int)(i % L1);
    float v = Xg[(LL)i0 * s0 + (LL)i1 * s1];
    s += v; q += v * v;
  }
  ss[t] = s; sq[t] = q; __syncthreads();
  for (int st = 128; st > 0; st >>= 1) {
    if (t < st) { ss[t] += ss[t + st]; sq[t] += sq[t + st]; }
    __syncthreads();
  }
  if (t == 0) {
    float m = ss[0] / (float)total;
    float v = sq[0] / (float)total - m * m;
    mu[g] = m;
    rstd[g] = rsqrtf(fmaxf(v, 0.f) + eps);
  }
}

__global__ void k_napply(const float* __restrict__ X, float* __restrict__ Y,
                         const float* __restrict__ mu, const float* __restrict__ rstd,
                         const float* __restrict__ gamma, const float* __restrict__ beta,
                         LL total, int statDiv, int statMod, int chDiv, int chMod,
                         int relu) {
  LL e = (LL)blockIdx.x * 256 + threadIdx.x;
  if (e >= total) return;
  int sIdx = (int)((e / statDiv) % statMod);
  int cIdx = (int)((e / chDiv) % chMod);
  float gm = gamma ? gamma[cIdx] : 1.f;
  float bt = beta ? beta[cIdx] : 0.f;
  float y = (X[e] - mu[sIdx]) * rstd[sIdx] * gm + bt;
  if (relu) y = fmaxf(y, 0.f);
  Y[e] = y;
}

__global__ void k_softrows(const float* __restrict__ S, float* __restrict__ P,
                           const float* __restrict__ mask, int N1, int N2, float scale) {
  const int g = blockIdx.y, row = blockIdx.x, t = threadIdx.x;
  const LL base = ((LL)g * N1 + row) * N2;
  const float* mrow = mask ? mask + (LL)(g / HEADS) * N2 : nullptr;
  __shared__ float red[256];
  float mx = -3.4e38f;
  for (int j = t; j < N2; j += 256) {
    float v = S[base + j];
    if (mrow && mrow[j] == 0.f) v = -1e6f;
    mx = fmaxf(mx, v * scale);
  }
  red[t] = mx; __syncthreads();
  for (int st = 128; st > 0; st >>= 1) { if (t < st) red[t] = fmaxf(red[t], red[t + st]); __syncthreads(); }
  mx = red[0]; __syncthreads();
  float sum = 0.f;
  for (int j = t; j < N2; j += 256) {
    float v = S[base + j];
    if (mrow && mrow[j] == 0.f) v = -1e6f;
    float e = __expf(v * scale - mx);
    P[base + j] = e; sum += e;
  }
  red[t] = sum; __syncthreads();
  for (int st = 128; st > 0; st >>= 1) { if (t < st) red[t] += red[t + st]; __syncthreads(); }
  float inv = 1.f / red[0];
  for (int j = t; j < N2; j += 256) P[base + j] *= inv;
}

__global__ void k_softcols(const float* __restrict__ S, float* __restrict__ Pc,
                           const float* __restrict__ mask, int n1, int n2, float scale) {
  const int g = blockIdx.y, col = blockIdx.x, t = threadIdx.x; // blockDim = n1 = 128
  __shared__ float red[128];
  const LL base = (LL)g * n1 * n2;
  float v = S[base + (LL)t * n2 + col];
  if (mask) { if (mask[(LL)(g / HEADS) * n2 + col] == 0.f) v = -1e6f; }
  v *= scale;
  red[t] = v; __syncthreads();
  for (int st = 64; st > 0; st >>= 1) { if (t < st) red[t] = fmaxf(red[t], red[t + st]); __syncthreads(); }
  float mx = red[0]; __syncthreads();
  float e = __expf(v - mx);
  red[t] = e; __syncthreads();
  for (int st = 64; st > 0; st >>= 1) { if (t < st) red[t] += red[t + st]; __syncthreads(); }
  Pc[base + (LL)t * n2 + col] = e / red[0];
}

__global__ void k_meanh(const float* __restrict__ Pm, float* __restrict__ cs, LL n1n2) {
  LL e = (LL)blockIdx.x * 256 + threadIdx.x;
  LL total = (LL)Bb * n1n2;
  if (e >= total) return;
  LL b = e / n1n2, r = e % n1n2;
  float a = 0.f;
  for (int h = 0; h < HEADS; ++h) a += Pm[((LL)b * HEADS + h) * n1n2 + r];
  cs[e] = a * (1.f / HEADS);
}

__global__ void k_meanhT(const float* __restrict__ Pc, float* __restrict__ ds,
                         int n1, int n2) {
  LL e = (LL)blockIdx.x * 256 + threadIdx.x;
  LL total = (LL)Bb * n1 * n2;
  if (e >= total) return;
  int i = (int)(e % n1);
  LL r = e / n1;
  int j = (int)(r % n2);
  int b = (int)(r / n2);
  float a = 0.f;
  for (int h = 0; h < HEADS; ++h) a += Pc[(((LL)b * HEADS + h) * n1 + i) * n2 + j];
  ds[e] = a * (1.f / HEADS);
}

// ---------------- misc elementwise ----------------
__global__ void k_mask(const float* __restrict__ lg, float* __restrict__ mk, LL n) {
  LL e = (LL)blockIdx.x * 256 + threadIdx.x;
  if (e < n) mk[e] = (lg[e] > 0.f) ? 1.f : 0.f;
}
__global__ void k_copy(float* __restrict__ dst, const float* __restrict__ src, LL n) {
  LL e = (LL)blockIdx.x * 256 + threadIdx.x;
  if (e < n) dst[e] = src[e];
}
__global__ void k_concat(float* __restrict__ dst, const float* __restrict__ a,
                         const float* __restrict__ b, int C1, int C2, int L) {
  LL e = (LL)blockIdx.x * 256 + threadIdx.x;
  LL total = (LL)Bb * (C1 + C2) * L;
  if (e >= total) return;
  int l = (int)(e % L);
  int c = (int)((e / L) % (C1 + C2));
  int g = (int)(e / ((LL)(C1 + C2) * L));
  dst[e] = (c < C1) ? a[((LL)g * C1 + c) * L + l] : b[((LL)g * C2 + (c - C1)) * L + l];
}

// ---------------- top-k / KNN ----------------
__global__ void k_topk(const float* __restrict__ sc, int* __restrict__ idxo,
                       int Ncols, int Ksel) {
  const int row = blockIdx.x, t = threadIdx.x;
  __shared__ float rowv[2048];
  __shared__ float rv[256];
  __shared__ int ri[256];
  const float* S = sc + (LL)row * Ncols;
  for (int j = t; j < 2048; j += 256) rowv[j] = (j < Ncols) ? S[j] : -3.4e38f;
  __syncthreads();
  for (int it = 0; it < Ksel; ++it) {
    float bv = -3.4e38f; int bi = 0x7fffffff;
    for (int j = t; j < Ncols; j += 256) {
      float v = rowv[j];
      if (v > bv || (v == bv && j < bi)) { bv = v; bi = j; }
    }
    rv[t] = bv; ri[t] = bi; __syncthreads();
    for (int st = 128; st > 0; st >>= 1) {
      if (t < st) {
        if (rv[t + st] > rv[t] || (rv[t + st] == rv[t] && ri[t + st] < ri[t])) {
          rv[t] = rv[t + st]; ri[t] = ri[t + st];
        }
      }
      __syncthreads();
    }
    if (t == 0) { idxo[(LL)row * Ksel + it] = ri[0]; rowv[ri[0]] = -3.4e38f; }
    __syncthreads();
  }
}

__global__ void k_gatherxt(const float* __restrict__ d, const int* __restrict__ idx,
                           float* __restrict__ xt) {
  LL e = (LL)blockIdx.x * 256 + threadIdx.x;
  LL total = (LL)Bb * MC * TOPM * Cc;
  if (e >= total) return;
  int c = (int)(e % Cc);
  LL r = e / Cc;
  int m = (int)(r % TOPM);
  int g = (int)(r / TOPM);
  int b = g >> 7;
  xt[e] = d[((LL)b * Cc + c) * Nn + idx[(LL)g * TOPM + m]];
}

__global__ void k_pdtopk(const float* __restrict__ gram, int* __restrict__ nn) {
  const int row = blockIdx.x;                 // g*TOPM + i
  const int g = row >> 7, i = row & 127, t = threadIdx.x;
  __shared__ float pd[128];
  __shared__ float rv[128];
  __shared__ int ri[128];
  const float* Gg = gram + (LL)g * TOPM * TOPM;
  float xi = Gg[i * TOPM + i];
  float xj = Gg[t * TOPM + t];
  pd[t] = 2.f * Gg[i * TOPM + t] - xi - xj;
  __syncthreads();
  for (int it = 0; it < KNNk; ++it) {
    rv[t] = pd[t]; ri[t] = t; __syncthreads();
    for (int st = 64; st > 0; st >>= 1) {
      if (t < st) {
        if (rv[t + st] > rv[t] || (rv[t + st] == rv[t] && ri[t + st] < ri[t])) {
          rv[t] = rv[t + st]; ri[t] = ri[t + st];
        }
      }
      __syncthreads();
    }
    if (t == 0) { nn[(LL)row * KNNk + it] = ri[0]; pd[ri[0]] = -3.4e38f; }
    __syncthreads();
  }
}

__global__ void k_featstats(const float* __restrict__ xt, const int* __restrict__ nn,
                            float* __restrict__ muf, float* __restrict__ varf) {
  const int row = blockIdx.x;            // g*2C + ch
  const int g = row >> 8, ch = row & 255, t = threadIdx.x;
  __shared__ float ss[128], sq[128];
  const float* xg = xt + (LL)g * TOPM * Cc;
  const int* ng = nn + (LL)g * TOPM * KNNk;
  float s = 0.f, q = 0.f;
  for (int e = t; e < TOPM * KNNk; e += 128) {
    int m = e >> 3, kq = e & 7;
    int chl = ch & (Cc - 1);
    float cen = xg[m * Cc + chl];
    float v = (ch < Cc) ? cen : (cen - xg[ng[m * KNNk + kq] * Cc + chl]);
    s += v; q += v * v;
  }
  ss[t] = s; sq[t] = q; __syncthreads();
  for (int st = 64; st > 0; st >>= 1) {
    if (t < st) { ss[t] += ss[t + st]; sq[t] += sq[t + st]; }
    __syncthreads();
  }
  if (t == 0) {
    float m = ss[0] / (float)(TOPM * KNNk);
    muf[row] = m;
    varf[row] = fmaxf(sq[0] / (float)(TOPM * KNNk) - m * m, 0.f);
  }
}

__global__ void k_bnvar(const float* __restrict__ varf, float* __restrict__ bnv) {
  const int ch = blockIdx.x, t = threadIdx.x;    // blockDim == 256 == #groups
  __shared__ float ss[256];
  float v = varf[(LL)t * 2 * Cc + ch];
  ss[t] = v / (v + IN_EPS);
  __syncthreads();
  for (int st = 128; st > 0; st >>= 1) { if (t < st) ss[t] += ss[t + st]; __syncthreads(); }
  if (t == 0) bnv[ch] = ss[0] / 256.f;
}

__global__ void k_mkscale(const float* __restrict__ muf, const float* __restrict__ varf,
                          const float* __restrict__ bnv, const float* __restrict__ gamma,
                          const float* __restrict__ beta, float* __restrict__ scl,
                          float* __restrict__ sh) {
  LL e = (LL)blockIdx.x * 256 + threadIdx.x;
  LL total = (LL)Bb * MC * 2 * Cc;
  if (e >= total) return;
  int ch = (int)(e & (2 * Cc - 1));
  float rin = rsqrtf(varf[e] + IN_EPS);
  float rbn = rsqrtf(bnv[ch] + BN_EPS);
  float a = gamma[ch] * rin * rbn;
  scl[e] = a;
  sh[e] = beta[ch] - muf[e] * a;
}

__global__ void k_scattermax(const float* __restrict__ dnew, const int* __restrict__ topidx,
                             float* __restrict__ dloc) {
  LL e = (LL)blockIdx.x * 256 + threadIdx.x;
  LL total = (LL)Bb * MC * Cc * TOPM;
  if (e >= total) return;
  int m = (int)(e % TOPM);
  LL r = e / TOPM;
  int c = (int)(r % Cc);
  int g = (int)(r / Cc);
  int b = g >> 7;
  int n = topidx[(LL)g * TOPM + m];
  atomicMaxF(&dloc[((LL)b * Cc + c) * Nn + n], dnew[e]);
}

// =====================================================================
// Host-side orchestration
// =====================================================================
static inline int gf(LL n) { return (int)((n + 255) / 256); }

static void gemm(hipStream_t s, const float* A, LL sAg, LL sAm, LL sAk,
                 const float* B, LL sBg, LL sBk, LL sBn,
                 float* D, LL sDg, LL sDm, LL sDn,
                 int M, int N, int K, int G,
                 const float* bias = nullptr, const float* res = nullptr,
                 LL sRg = 0, LL sRm = 0, LL sRn = 0, int relu = 0) {
  GemmP p;
  p.A = A; p.B = B; p.D = D; p.bias = bias; p.res = res;
  p.sAg = sAg; p.sAm = sAm; p.sAk = sAk;
  p.sBg = sBg; p.sBk = sBk; p.sBn = sBn;
  p.sDg = sDg; p.sDm = sDm; p.sDn = sDn;
  p.sRg = sRg; p.sRm = sRm; p.sRn = sRn;
  p.M = M; p.N = N; p.K = K; p.relu = relu;
  dim3 grid((N + 15) / 16, (M + 63) / 64, G);
  k_gemm<<<grid, 128, 0, s>>>(p);
}

struct AttnW {
  const float *Wq, *bq, *Wk, *bk, *Wv, *bv, *Wmh, *bmh, *Wc1, *bc1, *g1, *be1, *Wc2, *bc2;
};

static void run_attn(hipStream_t s, const AttnW& w, const float* x1, const float* x2,
                     int n1, int n2, const float* mask, float* outx,
                     float* q, float* k, float* v, float* S, float* Pm, float* att,
                     float* cat, float* hbuf, float* mu, float* rs,
                     float* colP, float* cs, float* ds) {
  const int G2 = Bb * HEADS;
  gemm(s, w.Wq, 0, Cc, 1, x1, (LL)Cc * n1, n1, 1, q, (LL)Cc * n1, n1, 1, Cc, n1, Cc, Bb, w.bq);
  gemm(s, w.Wk, 0, Cc, 1, x2, (LL)Cc * n2, n2, 1, k, (LL)Cc * n2, n2, 1, Cc, n2, Cc, Bb, w.bk);
  gemm(s, w.Wv, 0, Cc, 1, x2, (LL)Cc * n2, n2, 1, v, (LL)Cc * n2, n2, 1, Cc, n2, Cc, Bb, w.bv);
  // S[g](n1,n2) = q^T k   (K = HD = 32: one WMMA per tile)
  gemm(s, q, (LL)HDim * n1, 1, n1, k, (LL)HDim * n2, n2, 1,
       S, (LL)n1 * n2, n2, 1, n1, n2, HDim, G2);
  k_softrows<<<dim3(n1, G2), 256, 0, s>>>(S, Pm, mask, n1, n2, SM_SCALE);
  if (cs) {
    k_softcols<<<dim3(n2, G2), 128, 0, s>>>(S, colP, mask, n1, n2, SM_SCALE);
    k_meanh<<<gf((LL)Bb * n1 * n2), 256, 0, s>>>(Pm, cs, (LL)n1 * n2);
    k_meanhT<<<gf((LL)Bb * n1 * n2), 256, 0, s>>>(colP, ds, n1, n2);
  }
  // att[g](hd,n1) = v * P^T
  gemm(s, v, (LL)HDim * n2, n2, 1, Pm, (LL)n1 * n2, 1, n2,
       att, (LL)HDim * n1, n1, 1, HDim, n1, n2, G2);
  // mh (reuse k buffer)
  gemm(s, w.Wmh, 0, Cc, 1, att, (LL)Cc * n1, n1, 1, k, (LL)Cc * n1, n1, 1, Cc, n1, Cc, Bb, w.bmh);
  k_concat<<<gf((LL)Bb * 2 * Cc * n1), 256, 0, s>>>(cat, x1, k, Cc, Cc, n1);
  gemm(s, w.Wc1, 0, 2 * Cc, 1, cat, (LL)2 * Cc * n1, n1, 1,
       hbuf, (LL)2 * Cc * n1, n1, 1, 2 * Cc, n1, 2 * Cc, Bb, w.bc1);
  // BN over (B, n1) per channel (2C)
  k_stats<<<2 * Cc, 256, 0, s>>>(hbuf, mu, rs, (LL)n1, Bb, (LL)2 * Cc * n1, n1, 1, BN_EPS);
  k_napply<<<gf((LL)Bb * 2 * Cc * n1), 256, 0, s>>>(hbuf, hbuf, mu, rs, w.g1, w.be1,
                                                    (LL)Bb * 2 * Cc * n1, n1, 2 * Cc, n1, 2 * Cc, 1);
  // x1n = x1 + conv(h)
  gemm(s, w.Wc2, 0, 2 * Cc, 1, hbuf, (LL)2 * Cc * n1, n1, 1,
       outx, (LL)Cc * n1, n1, 1, Cc, n1, 2 * Cc, Bb, w.bc2,
       x1, (LL)Cc * n1, n1, 1, 0);
}

extern "C" void kernel_launch(void* const* d_in, const int* in_sizes, int n_in,
                              void* d_out, int out_size, void* d_ws, size_t ws_size,
                              hipStream_t s) {
  if (n_in < 71) return;
  (void)in_sizes; (void)out_size; (void)ws_size;
  auto P = [&](int i) { return (const float*)d_in[i]; };
  const float* din = P(1);
  const float* fpin = P(2);
  const float* logits = P(3);
  const float* cscore = P(4);
  const int A0 = 5, A1 = 19, A2 = 33, LC = 47, MG = 51, IL = 57;
  auto attw = [&](int b) {
    AttnW w;
    w.Wq = P(b + 0);  w.bq = P(b + 1);  w.Wk = P(b + 2);  w.bk = P(b + 3);
    w.Wv = P(b + 4);  w.bv = P(b + 5);  w.Wmh = P(b + 6); w.bmh = P(b + 7);
    w.Wc1 = P(b + 8); w.bc1 = P(b + 9); w.g1 = P(b + 10); w.be1 = P(b + 11);
    w.Wc2 = P(b + 12); w.bc2 = P(b + 13);
    return w;
  };

  // output layout (flat, return order)
  float* out = (float*)d_out;
  float* o_dm = out;                                       // [B,C,N]
  float* o_fp = out + (LL)Bb * Cc * Nn;                    // [B,C,128]
  float* o_lg = o_fp + (LL)Bb * Cc * MC;                   // [B,N]
  float* o_cs = o_lg + (LL)Bb * Nn;                        // [B,128,N]
  float* o_ds = o_cs + (LL)Bb * MC * Nn;                   // [B,N,128]

  // workspace bump allocator (floats)
  float* w = (float*)d_ws;
  size_t off = 0;
  auto A_ = [&](size_t n) { float* p = w + off; off += n; return p; };
  float* mask = A_((LL)Bb * Nn);
  float* qb   = A_((LL)Bb * Cc * Nn);
  float* kb   = A_((LL)Bb * Cc * Nn);
  float* vb   = A_((LL)Bb * Cc * Nn);
  float* Sb   = A_((LL)Bb * HEADS * Nn * MC);
  float* Pb   = A_((LL)Bb * HEADS * Nn * MC);
  float* colP = A_((LL)Bb * HEADS * Nn * MC);
  float* attb = A_((LL)Bb * Cc * Nn);
  float* catb = A_((LL)Bb * 2 * Cc * Nn);
  float* hb   = A_((LL)Bb * 2 * Cc * Nn);
  float* mu   = A_(65536);
  float* rs   = A_(65536);
  float* fp1  = A_((LL)Bb * Cc * MC);
  float* fp2  = A_((LL)Bb * Cc * MC);
  float* d2   = A_((LL)Bb * Cc * Nn);
  float* dloc = A_((LL)Bb * Cc * Nn);
  int*   topidx = (int*)A_((LL)Bb * MC * TOPM);
  int*   nnidx  = (int*)A_((LL)Bb * MC * TOPM * KNNk);
  float* xtb  = A_((LL)Bb * MC * TOPM * Cc);
  float* gram = A_((LL)Bb * MC * TOPM * TOPM);
  float* muf  = A_((LL)Bb * MC * 2 * Cc);
  float* varf = A_((LL)Bb * MC * 2 * Cc);
  float* bnv  = A_(2 * Cc);
  float* sclb = A_((LL)Bb * MC * 2 * Cc);
  float* shb  = A_((LL)Bb * MC * 2 * Cc);
  float* dnew = A_((LL)Bb * MC * Cc * TOPM);

  // ---- mask = relu(tanh(logits)) > 0 ----
  k_mask<<<gf((LL)Bb * Nn), 256, 0, s>>>(logits, mask, (LL)Bb * Nn);

  // ---- local_context(d, cluster_score) ----
  k_topk<<<Bb * MC, 256, 0, s>>>(cscore, topidx, Nn, TOPM);
  k_gatherxt<<<gf((LL)Bb * MC * TOPM * Cc), 256, 0, s>>>(din, topidx, xtb);
  // Gram = xt xt^T  (per g)
  gemm(s, xtb, (LL)TOPM * Cc, Cc, 1, xtb, (LL)TOPM * Cc, 1, Cc,
       gram, (LL)TOPM * TOPM, TOPM, 1, TOPM, TOPM, Cc, Bb * MC);
  k_pdtopk<<<Bb * MC * TOPM, 128, 0, s>>>(gram, nnidx);
  k_featstats<<<Bb * MC * 2 * Cc, 128, 0, s>>>(xtb, nnidx, muf, varf);
  k_bnvar<<<2 * Cc, 256, 0, s>>>(varf, bnv);
  k_mkscale<<<gf((LL)Bb * MC * 2 * Cc), 256, 0, s>>>(muf, varf, bnv, P(LC + 0), P(LC + 1), sclb, shb);
  k_localconv<<<dim3(TOPM * KNNk / 16, Cc / 16, Bb * MC), 32, 0, s>>>(
      P(LC + 2), P(LC + 3), xtb, nnidx, sclb, shb, dnew);
  k_copy<<<gf((LL)Bb * Cc * Nn), 256, 0, s>>>(dloc, din, (LL)Bb * Cc * Nn);
  k_scattermax<<<gf((LL)Bb * MC * Cc * TOPM), 256, 0, s>>>(dnew, topidx, dloc);

  // ---- attn0: feat_piece attends to d (masked), local scores ----
  AttnW w0 = attw(A0);
  run_attn(s, w0, fpin, din, MC, Nn, mask, fp1,
           qb, kb, vb, Sb, Pb, attb, catb, hb, mu, rs, colP, o_cs, o_ds);
  // ---- attn1: fp self-attention ----
  AttnW w1 = attw(A1);
  run_attn(s, w1, fp1, fp1, MC, MC, nullptr, fp2,
           qb, kb, vb, Sb, Pb, attb, catb, hb, mu, rs, nullptr, nullptr, nullptr);
  k_copy<<<gf((LL)Bb * Cc * MC), 256, 0, s>>>(o_fp, fp2, (LL)Bb * Cc * MC);
  // ---- attn2: d attends to fp ----
  AttnW w2 = attw(A2);
  run_attn(s, w2, din, fp2, Nn, MC, nullptr, d2,
           qb, kb, vb, Sb, Pb, attb, catb, hb, mu, rs, nullptr, nullptr, nullptr);

  // ---- merge: conv -> IN -> BN+relu -> conv ----
  k_concat<<<gf((LL)Bb * 2 * Cc * Nn), 256, 0, s>>>(catb, d2, dloc, Cc, Cc, Nn);
  gemm(s, P(MG + 0), 0, 2 * Cc, 1, catb, (LL)2 * Cc * Nn, Nn, 1,
       hb, (LL)Cc * Nn, Nn, 1, Cc, Nn, 2 * Cc, Bb, P(MG + 1));
  k_stats<<<Bb * Cc, 256, 0, s>>>(hb, mu, rs, (LL)Nn, 1, 0, Nn, 1, IN_EPS);
  k_napply<<<gf((LL)Bb * Cc * Nn), 256, 0, s>>>(hb, hb, mu, rs, nullptr, nullptr,
                                                (LL)Bb * Cc * Nn, Nn, Bb * Cc, Nn, Cc, 0);
  k_stats<<<Cc, 256, 0, s>>>(hb, mu, rs, (LL)Nn, Bb, (LL)Cc * Nn, Nn, 1, BN_EPS);
  k_napply<<<gf((LL)Bb * Cc * Nn), 256, 0, s>>>(hb, hb, mu, rs, P(MG + 2), P(MG + 3),
                                                (LL)Bb * Cc * Nn, Nn, Cc, Nn, Cc, 1);
  gemm(s, P(MG + 4), 0, Cc, 1, hb, (LL)Cc * Nn, Nn, 1,
       o_dm, (LL)Cc * Nn, Nn, 1, Cc, Nn, Cc, Bb, P(MG + 5));

  // ---- inlier head: 3x (conv -> IN -> BN+relu) -> conv ----
  int dims[4] = {Cc, 64, 16, 4};
  const float* cur = o_dm;
  float* tb[2] = {qb, kb};
  for (int st = 0; st < 3; ++st) {
    int Ci = dims[st], Co = dims[st + 1];
    float* tt = tb[st & 1];
    gemm(s, P(IL + st * 4 + 0), 0, Ci, 1, cur, (LL)Ci * Nn, Nn, 1,
         tt, (LL)Co * Nn, Nn, 1, Co, Nn, Ci, Bb, P(IL + st * 4 + 1));
    k_stats<<<Bb * Co, 256, 0, s>>>(tt, mu, rs, (LL)Nn, 1, 0, Nn, 1, IN_EPS);
    k_napply<<<gf((LL)Bb * Co * Nn), 256, 0, s>>>(tt, tt, mu, rs, nullptr, nullptr,
                                                  (LL)Bb * Co * Nn, Nn, Bb * Co, Nn, Co, 0);
    k_stats<<<Co, 256, 0, s>>>(tt, mu, rs, (LL)Nn, Bb, (LL)Co * Nn, Nn, 1, BN_EPS);
    k_napply<<<gf((LL)Bb * Co * Nn), 256, 0, s>>>(tt, tt, mu, rs, P(IL + st * 4 + 2),
                                                  P(IL + st * 4 + 3),
                                                  (LL)Bb * Co * Nn, Nn, Co, Nn, Co, 1);
    cur = tt;
  }
  gemm(s, P(IL + 12), 0, 4, 1, cur, (LL)4 * Nn, Nn, 1,
       o_lg, (LL)Nn, Nn, 1, 1, Nn, 4, Bb, P(IL + 13));
}
